// EncoderBlock_7464653160895
// MI455X (gfx1250) — matrix operand order repaired and checked by reading, not compiled, not run on hardware
//
#include <hip/hip_runtime.h>
#include <hip/hip_bf16.h>

typedef _Float16 half_t;
typedef __attribute__((ext_vector_type(16))) _Float16 v16h;
typedef __attribute__((ext_vector_type(8)))  float    v8f;

#define B_  4
#define T_  2048
#define D_  512
#define H_  8
#define HD_ 64
#define MROWS (B_*T_)     // 8192

#define USE_ASYNC 1

static __device__ inline v8f zero8() {
    v8f z = {0.f,0.f,0.f,0.f,0.f,0.f,0.f,0.f};
    return z;
}

static __device__ inline v8f wmma_f16(const v16h& a, const v16h& b, const v8f& c) {
    return __builtin_amdgcn_wmma_f32_16x16x32_f16(false, a, false, b, (short)0, c, false, false);
}

// 16-byte global -> LDS copy. CDNA5 path: async DMA to LDS, tracked by ASYNCcnt.
static __device__ inline void async_copy16(half_t* lds, const half_t* g) {
#if USE_ASYNC
    unsigned int loff = (unsigned int)(unsigned long long)lds;   // low 32b = LDS offset
    asm volatile("global_load_async_to_lds_b128 %0, %1, off"
                 :: "v"(loff), "v"(g) : "memory");
#else
    *(uint4*)lds = *(const uint4*)g;
#endif
}

template<int N>
static __device__ inline void async_wait() {
#if USE_ASYNC
    asm volatile("s_wait_asynccnt %0" :: "i"(N) : "memory");
#endif
}

// ---------------------------------------------------------------------------
// Weight conversion: f32 -> f16, stored TRANSPOSED [N][K].
//   mode 0: plain W[K][N] -> Wt[N][K]          (ksh = log2(K), N given)
//   mode 1: Wqkv [H][D][64] -> Wt[H*64][D]     (N=512, K=512)
// ---------------------------------------------------------------------------
__global__ void convert_kernel(const float* __restrict__ in, half_t* __restrict__ out,
                               int total, int ksh, int N, int mode) {
    int i = blockIdx.x * 256 + threadIdx.x;
    if (i >= total) return;
    if (mode == 0) {
        int n = i >> ksh;
        int k = i & ((1 << ksh) - 1);
        out[i] = (half_t)in[(size_t)k * N + n];
    } else {
        int n = i >> 9;          // h*64+e
        int k = i & 511;         // d
        int h = n >> 6, e = n & 63;
        out[i] = (half_t)in[h * (D_*HD_) + k * HD_ + e];
    }
}

// ---------------------------------------------------------------------------
// LayerNorm over last dim (512), writes f16. One block (256 thr) per row.
// ---------------------------------------------------------------------------
__global__ __launch_bounds__(256) void ln_kernel(const float* __restrict__ x,
                                                 const float* __restrict__ g,
                                                 const float* __restrict__ b,
                                                 half_t* __restrict__ out) {
    __shared__ float red[16];
    int row = blockIdx.x;
    int tid = threadIdx.x;
    const float* xr = x + (size_t)row * D_;
    float x0 = xr[tid], x1 = xr[tid + 256];
    float s  = x0 + x1;
    float sq = x0 * x0 + x1 * x1;
    for (int d = 1; d < 32; d <<= 1) {
        s  += __shfl_xor(s,  d);
        sq += __shfl_xor(sq, d);
    }
    int wave = tid >> 5;
    if ((tid & 31) == 0) { red[wave] = s; red[8 + wave] = sq; }
    __syncthreads();
    float S = 0.f, SQ = 0.f;
    for (int i = 0; i < 8; ++i) { S += red[i]; SQ += red[8 + i]; }
    float mu  = S * (1.f / D_);
    float var = SQ * (1.f / D_) - mu * mu;
    float rs  = rsqrtf(var + 1e-5f);
    half_t* orow = out + (size_t)row * D_;
    orow[tid]       = (half_t)((x0 - mu) * rs * g[tid]       + b[tid]);
    orow[tid + 256] = (half_t)((x1 - mu) * rs * g[tid + 256] + b[tid + 256]);
}

// ---------------------------------------------------------------------------
// WMMA GEMM: C[M,N] = A[M,K](f16,row) @ Bt[N,K](f16,row, i.e. W^T) + bias.
// Block tile 128x128, 4 waves (128 thr), wave tile 64x64 (4x4 C-tiles),
// K-step 32. Async double-buffered LDS staging; 16 contiguous b128 fragment
// loads feed 16 WMMAs per wave per k-step.
// epi 0: QKV  -> f16 scattered to [B,H,T,64]
// epi 1: GELU -> f16 [M,N]
// epi 2: +resid -> f32 [M,N]
// epi 3: QKV-transposed (for V) -> f16 scattered to [B,H,64,T]
// ---------------------------------------------------------------------------
#define EPI_QKV  0
#define EPI_GELU 1
#define EPI_RES  2
#define EPI_QKVT 3

__global__ __launch_bounds__(128) void gemm_kernel(const half_t* __restrict__ A,
                                                   const half_t* __restrict__ Bt,
                                                   const float* __restrict__ bias,
                                                   const float* __restrict__ resid,
                                                   float* __restrict__ outF,
                                                   half_t* __restrict__ outH,
                                                   int M, int N, int K, int epi) {
    __shared__ half_t As[2][128 * 40];   // [m][k], pad stride 40 halves
    __shared__ half_t Bs[2][128 * 40];   // [n][k], pad stride 40 halves

    int tid  = threadIdx.x;
    int lane = tid & 31;
    int wave = tid >> 5;       // 0..3
    int wm = wave & 1;         // 64-row sub-block
    int wn = wave >> 1;        // 64-col sub-block
    int bm = blockIdx.y * 128;
    int bn = blockIdx.x * 128;
    int l15 = lane & 15;
    int lh  = lane >> 4;

    auto issue_tile = [&](int k0, int bufi) {
        #pragma unroll
        for (int i = 0; i < 4; ++i) {
            int c   = tid + i * 128;
            int row = c >> 2;
            int off = (c & 3) * 8;
            async_copy16(&As[bufi][row * 40 + off], &A [(size_t)(bm + row) * K + k0 + off]);
            async_copy16(&Bs[bufi][row * 40 + off], &Bt[(size_t)(bn + row) * K + k0 + off]);
        }
    };

    v8f acc[4][4];
    for (int sm = 0; sm < 4; ++sm)
        for (int sn = 0; sn < 4; ++sn)
            acc[sm][sn] = zero8();

    int nk = K >> 5;
    issue_tile(0, 0);
    for (int kb = 0; kb < nk; ++kb) {
        int bufi = kb & 1;
        if (kb + 1 < nk) { issue_tile((kb + 1) << 5, bufi ^ 1); async_wait<8>(); }
        else             { async_wait<0>(); }
        __syncthreads();

        v16h afrag[4];
        #pragma unroll
        for (int sm = 0; sm < 4; ++sm) {
            int m = wm * 64 + sm * 16 + l15;
            #pragma unroll
            for (int v = 0; v < 8; ++v) {
                int kk = ((v < 4) ? 0 : 16) + lh * 8 + (v & 3) * 2;
                afrag[sm][2 * v]     = As[bufi][m * 40 + kk];
                afrag[sm][2 * v + 1] = As[bufi][m * 40 + kk + 1];
            }
        }
        v16h bfrag[4];
        #pragma unroll
        for (int sn = 0; sn < 4; ++sn) {
            int n = wn * 64 + sn * 16 + l15;
            #pragma unroll
            for (int v = 0; v < 8; ++v) {
                int kk = ((v < 4) ? 0 : 16) + lh * 8 + (v & 3) * 2;
                bfrag[sn][2 * v]     = Bs[bufi][n * 40 + kk];
                bfrag[sn][2 * v + 1] = Bs[bufi][n * 40 + kk + 1];
            }
        }
        #pragma unroll
        for (int sm = 0; sm < 4; ++sm)
            #pragma unroll
            for (int sn = 0; sn < 4; ++sn)
                acc[sm][sn] = wmma_f16(afrag[sm], bfrag[sn], acc[sm][sn]);
        __syncthreads();
    }

    // Epilogue
    for (int sm = 0; sm < 4; ++sm)
        for (int sn = 0; sn < 4; ++sn)
            for (int v = 0; v < 8; ++v) {
                int gm = bm + wm * 64 + sm * 16 + v + lh * 8;
                int gn = bn + wn * 64 + sn * 16 + l15;
                float val = acc[sm][sn][v] + bias[gn];
                if (epi == EPI_QKV) {
                    int bb = gm >> 11, t = gm & 2047;
                    int hh = gn >> 6,  e = gn & 63;
                    outH[(((size_t)(bb * H_ + hh)) * T_ + t) * HD_ + e] = (half_t)val;
                } else if (epi == EPI_QKVT) {
                    int bb = gm >> 11, t = gm & 2047;
                    int hh = gn >> 6,  e = gn & 63;
                    outH[((size_t)(bb * H_ + hh) * HD_ + e) * T_ + t] = (half_t)val;
                } else if (epi == EPI_GELU) {
                    float gv = 0.5f * val * (1.f + erff(val * 0.70710678118654752f));
                    outH[(size_t)gm * N + gn] = (half_t)gv;
                } else {
                    outF[(size_t)gm * N + gn] = val + resid[(size_t)gm * N + gn];
                }
            }
}

// ---------------------------------------------------------------------------
// Flash attention. Q,K f16 [B,H,T,64]; V f16 TRANSPOSED [B,H,64,T].
// One wave per 16-query tile; 8 waves/block share async double-buffered
// 32-token K and V^T chunks. Output f16 [B*T, 512] (heads concatenated).
// ---------------------------------------------------------------------------
__global__ __launch_bounds__(256) void attn_kernel(const half_t* __restrict__ Q,
                                                   const half_t* __restrict__ Kg,
                                                   const half_t* __restrict__ Vt,
                                                   half_t* __restrict__ attn) {
    __shared__ half_t Ks [2][32 * 64];   // [tok][e]
    __shared__ half_t Vst[2][64 * 32];   // [e][tok]
    __shared__ half_t Ps [8 * 16 * 32];  // per-wave 16x32 P tile

    int tid  = threadIdx.x;
    int lane = tid & 31;
    int wave = tid >> 5;
    int bh   = blockIdx.x >> 4;          // b*H + h
    int qblk = blockIdx.x & 15;
    int q0   = qblk * 128 + wave * 16;
    int b    = bh >> 3;
    int h    = bh & 7;
    int l15  = lane & 15;
    int lh   = lane >> 4;
    size_t base = (size_t)bh * T_ * HD_;   // same footprint for K and V^T

    auto issue_chunk = [&](int c, int bufi) {
        int tok = tid >> 3, offk = (tid & 7) * 8;
        async_copy16(&Ks[bufi][tok * 64 + offk],
                     &Kg[base + (size_t)(c * 32 + tok) * HD_ + offk]);
        int e = tid >> 2, offv = (tid & 3) * 8;
        async_copy16(&Vst[bufi][e * 32 + offv],
                     &Vt[base + (size_t)e * T_ + c * 32 + offv]);
    };

    // Q fragments (16x64 as two 16x32 A-frags), kept in registers
    v16h qf[2];
    {
        const half_t* qrow = Q + base + (size_t)(q0 + l15) * HD_;
        #pragma unroll
        for (int v = 0; v < 8; ++v) {
            int e = ((v < 4) ? 0 : 16) + lh * 8 + (v & 3) * 2;
            qf[0][2 * v]     = qrow[e];
            qf[0][2 * v + 1] = qrow[e + 1];
            qf[1][2 * v]     = qrow[e + 32];
            qf[1][2 * v + 1] = qrow[e + 33];
        }
    }

    float m_i[8], l_i[8];
    v8f O[4];
    #pragma unroll
    for (int v = 0; v < 8; ++v) { m_i[v] = -3.0e38f; l_i[v] = 0.f; }
    #pragma unroll
    for (int c4 = 0; c4 < 4; ++c4) O[c4] = zero8();

    half_t* Pw = Ps + wave * 512;
    const int NC = T_ / 32;

    issue_chunk(0, 0);
    for (int c = 0; c < NC; ++c) {
        int bufi = c & 1;
        if (c + 1 < NC) { issue_chunk(c + 1, bufi ^ 1); async_wait<2>(); }
        else            { async_wait<0>(); }
        __syncthreads();

        // scores: two 16x16 token sub-tiles, K-dim = 64 (2 WMMAs each)
        v8f S[2];
        #pragma unroll
        for (int st = 0; st < 2; ++st) {
            v8f z = zero8();
            #pragma unroll
            for (int eh = 0; eh < 2; ++eh) {
                v16h bf;
                int tok = st * 16 + l15;
                #pragma unroll
                for (int v = 0; v < 8; ++v) {
                    int e = eh * 32 + ((v < 4) ? 0 : 16) + lh * 8 + (v & 3) * 2;
                    bf[2 * v]     = Ks[bufi][tok * 64 + e];
                    bf[2 * v + 1] = Ks[bufi][tok * 64 + e + 1];
                }
                z = wmma_f16(qf[eh], bf, z);
            }
            S[st] = z;
        }

        // online softmax in C-tile layout (row r = v + 8*lh, col = l15)
        float scale[8];
        #pragma unroll
        for (int v = 0; v < 8; ++v) {
            float s0 = S[0][v] * 0.125f;
            float s1 = S[1][v] * 0.125f;
            float mx = fmaxf(s0, s1);
            for (int d = 1; d < 16; d <<= 1) mx = fmaxf(mx, __shfl_xor(mx, d));
            float mn = fmaxf(m_i[v], mx);
            float sc = __expf(m_i[v] - mn);
            float p0 = __expf(s0 - mn);
            float p1 = __expf(s1 - mn);
            float sum = p0 + p1;
            for (int d = 1; d < 16; d <<= 1) sum += __shfl_xor(sum, d);
            l_i[v]  = l_i[v] * sc + sum;
            m_i[v]  = mn;
            scale[v] = sc;
            int r = v + lh * 8;
            Pw[r * 32 + l15]      = (half_t)p0;
            Pw[r * 32 + 16 + l15] = (half_t)p1;
        }
        #pragma unroll
        for (int c4 = 0; c4 < 4; ++c4)
            #pragma unroll
            for (int v = 0; v < 8; ++v)
                O[c4][v] *= scale[v];

        // P A-fragment (16x32) from per-wave LDS transpose
        v16h pf;
        #pragma unroll
        for (int v = 0; v < 8; ++v) {
            int tk = ((v < 4) ? 0 : 16) + lh * 8 + (v & 3) * 2;
            pf[2 * v]     = Pw[l15 * 32 + tk];
            pf[2 * v + 1] = Pw[l15 * 32 + tk + 1];
        }
        // AV: O[16 x 64] += P[16x32] @ V[32x64]  (V staged transposed)
        #pragma unroll
        for (int c4 = 0; c4 < 4; ++c4) {
            v16h vf;
            int e = c4 * 16 + l15;
            #pragma unroll
            for (int v = 0; v < 8; ++v) {
                int tk = ((v < 4) ? 0 : 16) + lh * 8 + (v & 3) * 2;
                vf[2 * v]     = Vst[bufi][e * 32 + tk];
                vf[2 * v + 1] = Vst[bufi][e * 32 + tk + 1];
            }
            O[c4] = wmma_f16(pf, vf, O[c4]);
        }
        __syncthreads();
    }

    // finalize: divide by row sums, store to [B*T, 512] (heads concatenated)
    #pragma unroll
    for (int v = 0; v < 8; ++v) {
        float inv = 1.f / l_i[v];
        int r = v + lh * 8;
        size_t row = (size_t)(b * T_ + q0 + r);
        #pragma unroll
        for (int c4 = 0; c4 < 4; ++c4) {
            attn[row * D_ + h * HD_ + c4 * 16 + l15] = (half_t)(O[c4][v] * inv);
        }
    }
}

// ---------------------------------------------------------------------------
// Host-side launch
// ---------------------------------------------------------------------------
extern "C" void kernel_launch(void* const* d_in, const int* in_sizes, int n_in,
                              void* d_out, int out_size, void* d_ws, size_t ws_size,
                              hipStream_t stream) {
    const float* x     = (const float*)d_in[0];
    const float* ln1_g = (const float*)d_in[1];
    const float* ln1_b = (const float*)d_in[2];
    const float* Wq    = (const float*)d_in[3];
    const float* bq    = (const float*)d_in[4];
    const float* Wk    = (const float*)d_in[5];
    const float* bk    = (const float*)d_in[6];
    const float* Wv    = (const float*)d_in[7];
    const float* bv    = (const float*)d_in[8];
    const float* Wo    = (const float*)d_in[9];
    const float* bo    = (const float*)d_in[10];
    const float* ln2_g = (const float*)d_in[11];
    const float* ln2_b = (const float*)d_in[12];
    const float* W1    = (const float*)d_in[13];
    const float* b1    = (const float*)d_in[14];
    const float* W2    = (const float*)d_in[15];
    const float* b2    = (const float*)d_in[16];
    float* out = (float*)d_out;

    char* ws = (char*)d_ws;
    size_t off = 0;
    auto alloc = [&](size_t bytes) { char* p = ws + off; off += (bytes + 255) & ~(size_t)255; return p; };

    half_t* xn   = (half_t*)alloc((size_t)MROWS * D_ * 2);       // ln1 out, reused for ln2 out
    half_t* wq_h = (half_t*)alloc((size_t)D_ * D_ * 2);          // all weights stored W^T [N][K]
    half_t* wk_h = (half_t*)alloc((size_t)D_ * D_ * 2);
    half_t* wv_h = (half_t*)alloc((size_t)D_ * D_ * 2);
    half_t* wo_h = (half_t*)alloc((size_t)D_ * D_ * 2);
    half_t* w1_h = (half_t*)alloc((size_t)D_ * 4 * D_ * 2);
    half_t* w2_h = (half_t*)alloc((size_t)4 * D_ * D_ * 2);
    half_t* q_h  = (half_t*)alloc((size_t)MROWS * D_ * 2);       // [B,H,T,64]
    half_t* k_h  = (half_t*)alloc((size_t)MROWS * D_ * 2);       // [B,H,T,64]
    half_t* vt_h = (half_t*)alloc((size_t)MROWS * D_ * 2);       // [B,H,64,T]
    half_t* at_h = (half_t*)alloc((size_t)MROWS * D_ * 2);
    float*  y1   = (float*)alloc((size_t)MROWS * D_ * 4);
    half_t* h1_h = (half_t*)alloc((size_t)MROWS * 4 * D_ * 2);
    (void)ws_size; (void)n_in; (void)in_sizes; (void)out_size;

    const int nW  = D_ * D_;        // 262144
    const int nW1 = D_ * 4 * D_;    // 1048576

    // weight conversion (all -> transposed f16)
    convert_kernel<<<nW  / 256, 256, 0, stream>>>(Wq, wq_h, nW, 9, 512, 1);
    convert_kernel<<<nW  / 256, 256, 0, stream>>>(Wk, wk_h, nW, 9, 512, 1);
    convert_kernel<<<nW  / 256, 256, 0, stream>>>(Wv, wv_h, nW, 9, 512, 1);
    convert_kernel<<<nW  / 256, 256, 0, stream>>>(Wo, wo_h, nW, 9, 512, 0);
    convert_kernel<<<nW1 / 256, 256, 0, stream>>>(W1, w1_h, nW1, 9, 2048, 0);   // K=512,N=2048
    convert_kernel<<<nW1 / 256, 256, 0, stream>>>(W2, w2_h, nW1, 11, 512, 0);   // K=2048,N=512

    // LN1
    ln_kernel<<<MROWS, 256, 0, stream>>>(x, ln1_g, ln1_b, xn);

    // QKV projections (V written transposed per-head)
    dim3 gD(D_ / 128, MROWS / 128);
    gemm_kernel<<<gD, 128, 0, stream>>>(xn, wq_h, bq, nullptr, nullptr, q_h,
                                        MROWS, D_, D_, EPI_QKV);
    gemm_kernel<<<gD, 128, 0, stream>>>(xn, wk_h, bk, nullptr, nullptr, k_h,
                                        MROWS, D_, D_, EPI_QKV);
    gemm_kernel<<<gD, 128, 0, stream>>>(xn, wv_h, bv, nullptr, nullptr, vt_h,
                                        MROWS, D_, D_, EPI_QKVT);

    // flash attention
    attn_kernel<<<B_ * H_ * (T_ / 128), 256, 0, stream>>>(q_h, k_h, vt_h, at_h);

    // output projection + residual
    gemm_kernel<<<gD, 128, 0, stream>>>(at_h, wo_h, bo, x, y1, nullptr,
                                        MROWS, D_, D_, EPI_RES);

    // LN2 (reuse xn buffer)
    ln_kernel<<<MROWS, 256, 0, stream>>>(y1, ln2_g, ln2_b, xn);

    // FFN
    dim3 gF1(4 * D_ / 128, MROWS / 128);
    gemm_kernel<<<gF1, 128, 0, stream>>>(xn, w1_h, b1, nullptr, nullptr, h1_h,
                                         MROWS, 4 * D_, D_, EPI_GELU);
    gemm_kernel<<<gD, 128, 0, stream>>>(h1_h, w2_h, b2, y1, out, nullptr,
                                        MROWS, D_, 4 * D_, EPI_RES);
}
